// CausalAttention_5119601017090
// MI455X (gfx1250) — matrix operand order repaired and checked by reading, not compiled, not run on hardware
//
#include <hip/hip_runtime.h>
#include <hip/hip_bf16.h>

// ---------------------------------------------------------------------------
// Problem constants
// ---------------------------------------------------------------------------
constexpr int Tn    = 2048;          // sequence length
constexpr int BATCH = 2;             // batch
constexpr int En    = 1024;          // embed dim
constexpr int Hn    = 16;            // heads
constexpr int Kd    = 64;            // head dim
constexpr int HK    = Hn * Kd;       // 1024
constexpr int Mrows = Tn * BATCH;    // 4096 flattened rows (m = t*B + b)
constexpr int CH    = 32;            // scan chunk (== WMMA K for bf16)
constexpr int NC    = Tn / CH;       // 64 sequential chunks

// ---------------------------------------------------------------------------
// WMMA types / helpers (gfx1250, wave32)
// ---------------------------------------------------------------------------
typedef __attribute__((ext_vector_type(16))) __bf16 v16bf;
typedef __attribute__((ext_vector_type(8)))  __bf16 bf16x8;
typedef __attribute__((ext_vector_type(8)))  float  v8f;

__device__ __forceinline__ v8f wmma_bf16(v16bf a, v16bf b, v8f c) {
  // D = A(16x32 bf16) * B(32x16 bf16) + C(16x16 f32)
  return __builtin_amdgcn_wmma_f32_16x16x32_bf16(
      /*neg_a=*/false, a, /*neg_b=*/false, b,
      /*c_mod=*/(short)0, c, /*reuse_a=*/false, /*reuse_b=*/false);
}

// A-matrix operand gather for a 16x32 bf16 tile:
// lane holds row M = lane&15; elements 0..15 = K[h*16..h*16+15],
// where h = lane>>4.
// Caller passes p = rowBase + h*16 (16B aligned); second half is at p+8.
__device__ __forceinline__ v16bf loadA(const __bf16* p) {
  bf16x8 lo = *(const bf16x8*)(p);
  bf16x8 hi = *(const bf16x8*)(p + 8);
  v16bf r;
#pragma unroll
  for (int i = 0; i < 8; ++i) { r[i] = lo[i]; r[8 + i] = hi[i]; }
  return r;
}

// ---------------------------------------------------------------------------
// K0: f32 -> bf16 conversion
// ---------------------------------------------------------------------------
__global__ void cvt_f32_bf16_k(const float* __restrict__ in,
                               __bf16* __restrict__ out, int n) {
  int i = blockIdx.x * blockDim.x + threadIdx.x;
  if (i < n) out[i] = (__bf16)in[i];
}

// ---------------------------------------------------------------------------
// K1: C[M,N] = A[M,K] @ W[N,K]^T + bias   (bf16 in, f32 out)
// One wave computes a 64x64 tile: 4x4 WMMA accumulators, K-loop step 32.
// Inner loop keeps only ONE B-operand live at a time and the kernel is
// pinned to 1 wave/EU so the 128-VGPR accumulator tile never spills.
// L2 (192MB) holds the whole working set; no LDS staging needed.
// ---------------------------------------------------------------------------
__global__ __launch_bounds__(32, 1)
void gemm_bf16_k(const __bf16* __restrict__ A,
                 const __bf16* __restrict__ W,
                 const float* __restrict__ bias,
                 float* __restrict__ C,
                 int Ndim, int Kdim) {
  const int lane = threadIdx.x & 31;
  const int h    = lane >> 4;
  const int l16  = lane & 15;
  const int tM   = blockIdx.x;   // 64-row tile
  const int tN   = blockIdx.y;   // 64-col tile

  const __bf16* aBase = A + (size_t)(tM * 64 + l16) * Kdim + h * 16;
  const __bf16* bBase = W + (size_t)(tN * 64 + l16) * Kdim + h * 16;

  v8f acc[4][4] = {};
  for (int k0 = 0; k0 < Kdim; k0 += 32) {
    v16bf a[4];
#pragma unroll
    for (int i = 0; i < 4; ++i) {
      const __bf16* ap = aBase + (size_t)(i * 16) * Kdim + k0;
      __builtin_prefetch(ap + 32, 0, 1);      // next k-step -> global_prefetch
      a[i] = loadA(ap);
    }
#pragma unroll
    for (int j = 0; j < 4; ++j) {
      // B operand: lane = column, 16 contiguous K values per lane-half.
      // Loaded per-j so at most one B tile is live alongside the 128-VGPR acc.
      const v16bf b = *(const v16bf*)(bBase + (size_t)(j * 16) * Kdim + k0);
#pragma unroll
      for (int i = 0; i < 4; ++i)
        acc[i][j] = wmma_bf16(a[i], b, acc[i][j]);
    }
  }
  // C/D layout: VGPR r -> row (r + 8*h), col = l16
#pragma unroll
  for (int i = 0; i < 4; ++i)
#pragma unroll
    for (int j = 0; j < 4; ++j) {
      const int col = tN * 64 + j * 16 + l16;
      const float bs = bias[col];
#pragma unroll
      for (int r = 0; r < 8; ++r) {
        const int row = tM * 64 + i * 16 + r + h * 8;
        C[(size_t)row * Ndim + col] = acc[i][j][r] + bs;
      }
    }
}

// ---------------------------------------------------------------------------
// K2: q-softmax-mean: qs[m, :] = rsqrt(t+1)/H * sum_h softmax(q[m,h,:])
// Block = 512 (16 waves); wave h does one head with shfl reductions;
// heads merge via ds_add_f32 shared atomics.
// ---------------------------------------------------------------------------
__global__ void q_softmax_mean_k(const float* __restrict__ qraw,
                                 __bf16* __restrict__ qs) {
  __shared__ float accum[Kd];
  const int m    = blockIdx.x;
  const int wave = threadIdx.x >> 5;
  const int lane = threadIdx.x & 31;
  if (threadIdx.x < Kd) accum[threadIdx.x] = 0.f;
  __syncthreads();

  const float* row = qraw + (size_t)m * HK + wave * Kd;
  float v0 = row[lane], v1 = row[lane + 32];
  float mx = fmaxf(v0, v1);
#pragma unroll
  for (int off = 16; off > 0; off >>= 1) mx = fmaxf(mx, __shfl_xor(mx, off, 32));
  float e0 = __expf(v0 - mx), e1 = __expf(v1 - mx);
  float s = e0 + e1;
#pragma unroll
  for (int off = 16; off > 0; off >>= 1) s += __shfl_xor(s, off, 32);
  const float inv = 1.f / s;
  atomicAdd(&accum[lane], e0 * inv);
  atomicAdd(&accum[lane + 32], e1 * inv);
  __syncthreads();

  if (threadIdx.x < Kd) {
    const int t = m >> 1;  // m = t*BATCH + b
    const float scale = rsqrtf((float)(t + 1)) * (1.f / (float)Hn);
    qs[(size_t)m * Kd + threadIdx.x] = (__bf16)(accum[threadIdx.x] * scale);
  }
}

// ---------------------------------------------------------------------------
// K3: LayerNorm over 64 (k path). One wave per row; writes row-major kn
// and transposed knT[b][feat][t] (for WMMA A-operand in the scan).
// ---------------------------------------------------------------------------
__global__ void ln_k_k(const float* __restrict__ kraw,
                       const float* __restrict__ gamma,
                       const float* __restrict__ beta,
                       __bf16* __restrict__ kn,
                       __bf16* __restrict__ knT) {
  const int m    = (blockIdx.x * blockDim.x + threadIdx.x) >> 5;
  const int lane = threadIdx.x & 31;
  const float v0 = kraw[(size_t)m * Kd + lane];
  const float v1 = kraw[(size_t)m * Kd + 32 + lane];
  float s = v0 + v1;
#pragma unroll
  for (int off = 16; off > 0; off >>= 1) s += __shfl_xor(s, off, 32);
  const float mu = s * (1.f / (float)Kd);
  const float d0 = v0 - mu, d1 = v1 - mu;
  float vs = d0 * d0 + d1 * d1;
#pragma unroll
  for (int off = 16; off > 0; off >>= 1) vs += __shfl_xor(vs, off, 32);
  const float inv = rsqrtf(vs * (1.f / (float)Kd) + 1e-5f);
  const float y0 = d0 * inv * gamma[lane] + beta[lane];
  const float y1 = d1 * inv * gamma[lane + 32] + beta[lane + 32];
  kn[(size_t)m * Kd + lane]      = (__bf16)y0;
  kn[(size_t)m * Kd + 32 + lane] = (__bf16)y1;
  const int t = m >> 1, b = m & 1;
  knT[((size_t)b * Kd + lane)      * Tn + t] = (__bf16)y0;
  knT[((size_t)b * Kd + 32 + lane) * Tn + t] = (__bf16)y1;
}

// ---------------------------------------------------------------------------
// K4: LayerNorm over 1024 (v path). One 256-thread block per row;
// writes transposed vnT[b][d][t] (WMMA B-operand in the scan).
// ---------------------------------------------------------------------------
__global__ void ln_v_k(const float* __restrict__ vraw,
                       const float* __restrict__ gamma,
                       const float* __restrict__ beta,
                       __bf16* __restrict__ vnT) {
  __shared__ float sbuf[256];
  const int m = blockIdx.x;
  const float* row = vraw + (size_t)m * En;
  float vals[4];
  float ls = 0.f;
#pragma unroll
  for (int i = 0; i < 4; ++i) { vals[i] = row[threadIdx.x + i * 256]; ls += vals[i]; }
  sbuf[threadIdx.x] = ls;
  __syncthreads();
  for (int st = 128; st > 0; st >>= 1) {
    if ((int)threadIdx.x < st) sbuf[threadIdx.x] += sbuf[threadIdx.x + st];
    __syncthreads();
  }
  const float mu = sbuf[0] * (1.f / (float)En);
  __syncthreads();
  float lv = 0.f;
#pragma unroll
  for (int i = 0; i < 4; ++i) { const float d = vals[i] - mu; lv += d * d; }
  sbuf[threadIdx.x] = lv;
  __syncthreads();
  for (int st = 128; st > 0; st >>= 1) {
    if ((int)threadIdx.x < st) sbuf[threadIdx.x] += sbuf[threadIdx.x + st];
    __syncthreads();
  }
  const float inv = rsqrtf(sbuf[0] * (1.f / (float)En) + 1e-5f);
  const int t = m >> 1, b = m & 1;
#pragma unroll
  for (int i = 0; i < 4; ++i) {
    const int d = threadIdx.x + i * 256;
    const float y = (vals[i] - mu) * inv * gamma[d] + beta[d];
    vnT[((size_t)b * En + d) * Tn + t] = (__bf16)y;
  }
}

// ---------------------------------------------------------------------------
// K5: intra-chunk attention A'[b,nc] = causal_mask( Qs_chunk @ Kn_chunk^T ),
// 32x32 per chunk, contraction over feature dim 64 (2 WMMA k-steps).
// kn rows serve directly as B-operand columns (feature-contiguous).
// ---------------------------------------------------------------------------
__global__ __launch_bounds__(32, 1)
void aprime_k(const __bf16* __restrict__ qs,
              const __bf16* __restrict__ kn,
              __bf16* __restrict__ Ap) {
  const int idx  = blockIdx.x;        // b*NC + nc
  const int b    = idx / NC;
  const int nc   = idx % NC;
  const int lane = threadIdx.x & 31;
  const int h    = lane >> 4;
  const int l16  = lane & 15;

  v8f acc[2][2] = {};
#pragma unroll
  for (int ks = 0; ks < 2; ++ks) {
    v16bf a[2];
#pragma unroll
    for (int i = 0; i < 2; ++i) {
      const int t = nc * CH + i * 16 + l16;
      const int m = t * BATCH + b;
      a[i] = loadA(qs + (size_t)m * Kd + ks * 32 + h * 16);
    }
#pragma unroll
    for (int j = 0; j < 2; ++j) {
      const int s = nc * CH + j * 16 + l16;
      const int ms = s * BATCH + b;
      const v16bf bb = *(const v16bf*)(kn + (size_t)ms * Kd + ks * 32 + h * 16);
#pragma unroll
      for (int i = 0; i < 2; ++i)
        acc[i][j] = wmma_bf16(a[i], bb, acc[i][j]);
    }
  }
  __bf16* out = Ap + (size_t)idx * (CH * CH);
#pragma unroll
  for (int i = 0; i < 2; ++i)
#pragma unroll
    for (int j = 0; j < 2; ++j) {
      const int sloc = j * 16 + l16;
#pragma unroll
      for (int r = 0; r < 8; ++r) {
        const int tloc = i * 16 + r + h * 8;
        const float v = (sloc <= tloc) ? acc[i][j][r] : 0.f;
        out[tloc * CH + sloc] = (__bf16)v;
      }
    }
}

// ---------------------------------------------------------------------------
// K6: chunked causal scan. One wave owns (b, 16-column E-tile); state
// S[64x16] lives in 4 WMMA accumulators, updated in place by S += K^T V.
// Per chunk: S mirrored to LDS as bf16 (B-operand for Qs@S), then
//   Y = Qs@S_prev + A'@Vc ;  S += Kc^T@Vc ;  store Y.
// 10 WMMAs / chunk / wave, 64 sequential chunks, 128 parallel waves.
// ---------------------------------------------------------------------------
__global__ __launch_bounds__(32, 1)
void scan_k(const __bf16* __restrict__ qs,
            const __bf16* __restrict__ knT,
            const __bf16* __restrict__ vnT,
            const __bf16* __restrict__ Ap,
            float* __restrict__ out) {
  __shared__ __bf16 Smir[16 * Kd];          // [col][k], 2KB
  const int idx   = blockIdx.x;             // b*(E/16) + dtile
  const int b     = idx / (En / 16);
  const int dtile = idx % (En / 16);
  const int lane  = threadIdx.x & 31;
  const int h     = lane >> 4;
  const int l16   = lane & 15;

  v8f S[4] = {};
  for (int nc = 0; nc < NC; ++nc) {
    // 1. bf16 mirror of S into LDS: Smir[col][k]
#pragma unroll
    for (int kb = 0; kb < 4; ++kb)
#pragma unroll
      for (int r = 0; r < 8; ++r) {
        const int krow = kb * 16 + r + h * 8;
        Smir[l16 * Kd + krow] = (__bf16)S[kb][r];
      }
    __syncthreads();  // single-wave WG: lowers to wait, not a real barrier

    v8f Y[2] = {};
    // 2. Y += Qs_chunk @ S_prev  (contraction over 64 S-rows, 2 k-steps)
#pragma unroll
    for (int ks = 0; ks < 2; ++ks) {
      const v16bf bS = *(const v16bf*)(&Smir[l16 * Kd + ks * 32 + h * 16]);
#pragma unroll
      for (int i = 0; i < 2; ++i) {
        const int t = nc * CH + i * 16 + l16;
        const int m = t * BATCH + b;
        const v16bf a = loadA(qs + (size_t)m * Kd + ks * 32 + h * 16);
        Y[i] = wmma_bf16(a, bS, Y[i]);
      }
    }
    // 3. Vc B-operand: vnT column-contiguous in t
    const v16bf bV = *(const v16bf*)(
        vnT + ((size_t)b * En + dtile * 16 + l16) * Tn + nc * CH + h * 16);
    // 4. Y += A' @ Vc (intra-chunk causal part)
    const __bf16* ap = Ap + (size_t)(b * NC + nc) * (CH * CH);
#pragma unroll
    for (int i = 0; i < 2; ++i) {
      const v16bf a = loadA(ap + (i * 16 + l16) * CH + h * 16);
      Y[i] = wmma_bf16(a, bV, Y[i]);
    }
    // 5. S += Kc^T @ Vc  (C-operand chaining keeps S in accumulators)
#pragma unroll
    for (int kb = 0; kb < 4; ++kb) {
      const v16bf a = loadA(knT + ((size_t)b * Kd + kb * 16 + l16) * Tn +
                            nc * CH + h * 16);
      S[kb] = wmma_bf16(a, bV, S[kb]);
    }
    // 6. store Y -> out[t, b, d]
#pragma unroll
    for (int i = 0; i < 2; ++i)
#pragma unroll
      for (int r = 0; r < 8; ++r) {
        const int t = nc * CH + i * 16 + r + h * 8;
        out[((size_t)t * BATCH + b) * En + dtile * 16 + l16] = Y[i][r];
      }
    __syncthreads();
  }
}

// ---------------------------------------------------------------------------
// Host-side launch
// ---------------------------------------------------------------------------
extern "C" void kernel_launch(void* const* d_in, const int* in_sizes, int n_in,
                              void* d_out, int out_size, void* d_ws, size_t ws_size,
                              hipStream_t stream) {
  const float* x      = (const float*)d_in[0];
  // d_in[1] = attn_mask (causal, handled analytically) -- unused
  const float* Wq     = (const float*)d_in[2];
  const float* bq     = (const float*)d_in[3];
  const float* Wk     = (const float*)d_in[4];
  const float* bk     = (const float*)d_in[5];
  const float* Wv     = (const float*)d_in[6];
  const float* bv     = (const float*)d_in[7];
  const float* kgam   = (const float*)d_in[8];
  const float* kbet   = (const float*)d_in[9];
  const float* vgam   = (const float*)d_in[10];
  const float* vbet   = (const float*)d_in[11];
  float* out = (float*)d_out;

  char* ws = (char*)d_ws;
  auto carve = [&](size_t bytes) -> char* {
    char* p = ws;
    ws += (bytes + 255) & ~(size_t)255;
    return p;
  };
  __bf16* xh   = (__bf16*)carve((size_t)Mrows * En * 2);
  __bf16* wqh  = (__bf16*)carve((size_t)HK * En * 2);
  __bf16* wkh  = (__bf16*)carve((size_t)Kd * En * 2);
  __bf16* wvh  = (__bf16*)carve((size_t)En * En * 2);
  float*  qraw = (float*)carve((size_t)Mrows * HK * 4);
  float*  kraw = (float*)carve((size_t)Mrows * Kd * 4);
  float*  vraw = (float*)carve((size_t)Mrows * En * 4);
  __bf16* qsb  = (__bf16*)carve((size_t)Mrows * Kd * 2);
  __bf16* kn   = (__bf16*)carve((size_t)Mrows * Kd * 2);
  __bf16* knT  = (__bf16*)carve((size_t)BATCH * Kd * Tn * 2);
  __bf16* vnT  = (__bf16*)carve((size_t)BATCH * En * Tn * 2);
  __bf16* Ap   = (__bf16*)carve((size_t)BATCH * NC * CH * CH * 2);

  // 0) f32 -> bf16 for GEMM operands
  {
    int n;
    n = Mrows * En; cvt_f32_bf16_k<<<(n + 255) / 256, 256, 0, stream>>>(x, xh, n);
    n = HK * En;    cvt_f32_bf16_k<<<(n + 255) / 256, 256, 0, stream>>>(Wq, wqh, n);
    n = Kd * En;    cvt_f32_bf16_k<<<(n + 255) / 256, 256, 0, stream>>>(Wk, wkh, n);
    n = En * En;    cvt_f32_bf16_k<<<(n + 255) / 256, 256, 0, stream>>>(Wv, wvh, n);
  }
  // 1) projections (WMMA bf16 GEMMs, f32 accumulate)
  gemm_bf16_k<<<dim3(Mrows / 64, HK / 64), 32, 0, stream>>>(xh, wqh, bq, qraw, HK, En);
  gemm_bf16_k<<<dim3(Mrows / 64, Kd / 64), 32, 0, stream>>>(xh, wkh, bk, kraw, Kd, En);
  gemm_bf16_k<<<dim3(Mrows / 64, En / 64), 32, 0, stream>>>(xh, wvh, bv, vraw, En, En);
  // 2) pointwise/reduction stages
  q_softmax_mean_k<<<Mrows, 512, 0, stream>>>(qraw, qsb);
  ln_k_k<<<Mrows / 8, 256, 0, stream>>>(kraw, kgam, kbet, kn, knT);
  ln_v_k<<<Mrows, 256, 0, stream>>>(vraw, vgam, vbet, vnT);
  // 3) intra-chunk causal attention matrices
  aprime_k<<<BATCH * NC, 32, 0, stream>>>(qsb, kn, Ap);
  // 4) chunked WMMA causal scan -> output
  scan_k<<<BATCH * (En / 16), 32, 0, stream>>>(qsb, knT, vnT, Ap, out);

  (void)in_sizes; (void)n_in; (void)out_size; (void)ws_size;
}
